// DeterministicMoERouter_60163901882949
// MI455X (gfx1250) — matrix operand order repaired and checked by reading, not compiled, not run on hardware
//
#include <hip/hip_runtime.h>
#include <hip/hip_bf16.h>

typedef __attribute__((ext_vector_type(16))) __bf16 v16bf;
typedef __attribute__((ext_vector_type(8)))  float  v8f;
typedef __attribute__((ext_vector_type(4)))  unsigned int v4u;

#define HID    2048
#define NEXP   128
#define TOPK   6
#define KSTEP  32
#define NSTEPS (HID / KSTEP)   // 64
#define BPAD   40              // padded per-expert k-extent in LDS (bf16): 80B row stride
#define ROWS_PER_BLOCK 128

__device__ __forceinline__ unsigned pack_bf16(__bf16 a, __bf16 b) {
  union { __bf16 h[2]; unsigned u; } t;
  t.h[0] = a; t.h[1] = b;
  return t.u;
}

__device__ __forceinline__ void split_bf16(float x, __bf16& hi, __bf16& lo) {
  hi = (__bf16)x;
  lo = (__bf16)(x - (float)hi);
}

// Async 16-byte copy global -> LDS (gfx1250). lds_off is the byte offset of the
// destination inside the workgroup's LDS allocation (low 32 bits of a generic
// pointer to a __shared__ object). Tracked by ASYNCcnt.
__device__ __forceinline__ void async_cp_b128(unsigned lds_off, const void* src) {
  asm volatile("global_load_async_to_lds_b128 %0, %1, off"
               :: "v"(lds_off), "v"(src)
               : "memory");
}

__device__ __forceinline__ unsigned lds_off_u32(const void* p) {
  return (unsigned)(size_t)p;   // generic LDS pointer: addr[31:0] = LDS offset
}

// ---------------------------------------------------------------------------
// Kernel 0: one-time gate split  W(fp32)[E,H] -> Whi,Wlo (bf16)[E,H] in d_ws
// ---------------------------------------------------------------------------
__global__ __launch_bounds__(256)
void split_gate(const float* __restrict__ W, __bf16* __restrict__ Whi,
                __bf16* __restrict__ Wlo) {
  const int i = (blockIdx.x * 256 + threadIdx.x) * 8;   // 8 elems / thread
  float4 a = *(const float4*)(W + i);
  float4 b = *(const float4*)(W + i + 4);
  __bf16 h[8], l[8];
  split_bf16(a.x, h[0], l[0]); split_bf16(a.y, h[1], l[1]);
  split_bf16(a.z, h[2], l[2]); split_bf16(a.w, h[3], l[3]);
  split_bf16(b.x, h[4], l[4]); split_bf16(b.y, h[5], l[5]);
  split_bf16(b.z, h[6], l[6]); split_bf16(b.w, h[7], l[7]);
  uint4 uh = {pack_bf16(h[0], h[1]), pack_bf16(h[2], h[3]),
              pack_bf16(h[4], h[5]), pack_bf16(h[6], h[7])};
  uint4 ul = {pack_bf16(l[0], l[1]), pack_bf16(l[2], l[3]),
              pack_bf16(l[4], l[5]), pack_bf16(l[6], l[7])};
  *(uint4*)(Whi + i) = uh;
  *(uint4*)(Wlo + i) = ul;
}

// ---------------------------------------------------------------------------
// Kernel 1: logits = X[N,H] * W[E,H]^T  via split-bf16 WMMA (3 products).
// ASYNC_WS=true : gate tiles double-buffered into LDS with
//                 global_load_async_to_lds_b128 (ASYNCcnt pipeline).
// ASYNC_WS=false: round-1 synchronous staging (fallback, no workspace).
// ---------------------------------------------------------------------------
template <bool ASYNC_WS>
__global__ __launch_bounds__(256)
void router_gemm(const float* __restrict__ X, const float* __restrict__ W,
                 const __bf16* __restrict__ Whi, const __bf16* __restrict__ Wlo,
                 float* __restrict__ logits) {
  __shared__ __align__(16) __bf16 Bhi[2][NEXP * BPAD];
  __shared__ __align__(16) __bf16 Blo[2][NEXP * BPAD];

  const int tid  = threadIdx.x;
  const int wave = tid >> 5;
  const int lane = tid & 31;
  const int hh   = lane >> 4;   // lane half (0/1)
  const int l16  = lane & 15;

  const int rowBase = blockIdx.x * ROWS_PER_BLOCK + wave * 16;
  const float* xrow = X + (size_t)(rowBase + l16) * HID;

  v8f acc[8];
#pragma unroll
  for (int s = 0; s < 8; ++s) acc[s] = (v8f){0.f, 0.f, 0.f, 0.f, 0.f, 0.f, 0.f, 0.f};

  // staging assignment: thread t -> expert e = t/2, k-range (t&1)*16..+16
  const int ge = tid >> 1;
  const int gk = (tid & 1) * 16;

  const __bf16* srcHi = Whi + (size_t)ge * HID + gk;
  const __bf16* srcLo = Wlo + (size_t)ge * HID + gk;
  const unsigned dH[2] = {lds_off_u32(&Bhi[0][ge * BPAD + gk]),
                          lds_off_u32(&Bhi[1][ge * BPAD + gk])};
  const unsigned dL[2] = {lds_off_u32(&Blo[0][ge * BPAD + gk]),
                          lds_off_u32(&Blo[1][ge * BPAD + gk])};

  if constexpr (ASYNC_WS) {
    // prologue: buffer 0 <- k-chunk 0 (4 x b128 per thread)
    async_cp_b128(dH[0], srcHi);      async_cp_b128(dH[0] + 16, srcHi + 8);
    async_cp_b128(dL[0], srcLo);      async_cp_b128(dL[0] + 16, srcLo + 8);
  }

  for (int step = 0; step < NSTEPS; ++step) {
    const int k0 = step * KSTEP;
    int cur;
    if constexpr (ASYNC_WS) {
      cur = step & 1;
      if (step + 1 < NSTEPS) {
        // kick off next chunk into the other buffer, then wait for current 4
        const __bf16* sh = srcHi + (size_t)(step + 1) * KSTEP;
        const __bf16* sl = srcLo + (size_t)(step + 1) * KSTEP;
        const unsigned dh = dH[cur ^ 1], dl = dL[cur ^ 1];
        async_cp_b128(dh, sh);        async_cp_b128(dh + 16, sh + 8);
        async_cp_b128(dl, sl);        async_cp_b128(dl + 16, sl + 8);
        asm volatile("s_wait_asynccnt 0x4" ::: "memory");
      } else {
        asm volatile("s_wait_asynccnt 0x0" ::: "memory");
      }
      __syncthreads();
    } else {
      cur = 0;
      const float* wsrc = W + (size_t)ge * HID + k0 + gk;
      unsigned* bhU = (unsigned*)&Bhi[0][ge * BPAD + gk];
      unsigned* blU = (unsigned*)&Blo[0][ge * BPAD + gk];
#pragma unroll
      for (int j = 0; j < 16; j += 4) {
        float4 v = *(const float4*)(wsrc + j);
        __bf16 h0, h1, h2, h3, l0, l1, l2, l3;
        split_bf16(v.x, h0, l0); split_bf16(v.y, h1, l1);
        split_bf16(v.z, h2, l2); split_bf16(v.w, h3, l3);
        bhU[j / 2 + 0] = pack_bf16(h0, h1);
        bhU[j / 2 + 1] = pack_bf16(h2, h3);
        blU[j / 2 + 0] = pack_bf16(l0, l1);
        blU[j / 2 + 1] = pack_bf16(l2, l3);
      }
      __syncthreads();
    }

    // ---- A fragments (16-bit A 16x32 layout): lane half hh covers
    //      K = 16*hh + [0,16) contiguous (matches B) ----
    v16bf a_hi, a_lo;
    {
      const float* p0 = xrow + k0 + hh * 16;
      const float* p1 = xrow + k0 + hh * 16 + 8;
      float4 q0 = *(const float4*)(p0);
      float4 q1 = *(const float4*)(p0 + 4);
      float4 q2 = *(const float4*)(p1);
      float4 q3 = *(const float4*)(p1 + 4);
      float xs[16] = {q0.x, q0.y, q0.z, q0.w, q1.x, q1.y, q1.z, q1.w,
                      q2.x, q2.y, q2.z, q2.w, q3.x, q3.y, q3.z, q3.w};
#pragma unroll
      for (int i = 0; i < 16; ++i) {
        __bf16 h, l;
        split_bf16(xs[i], h, l);
        a_hi[i] = h;
        a_lo[i] = l;
      }
    }

    // ---- 8 expert subtiles; B 32x16 layout: lane l16 = expert column,
    //      lanes 0-15 hold K=0..15, lanes 16-31 hold K=16..31 ----
#pragma unroll
    for (int s = 0; s < 8; ++s) {
      const int e = s * 16 + l16;
      const __bf16* bh = &Bhi[cur][e * BPAD + hh * 16];
      const __bf16* bl = &Blo[cur][e * BPAD + hh * 16];
      union { v4u u[2]; v16bf v; } tb_hi, tb_lo;
      tb_hi.u[0] = *(const v4u*)(bh);
      tb_hi.u[1] = *(const v4u*)(bh + 8);
      tb_lo.u[0] = *(const v4u*)(bl);
      tb_lo.u[1] = *(const v4u*)(bl + 8);

      acc[s] = __builtin_amdgcn_wmma_f32_16x16x32_bf16(
          false, a_hi, false, tb_hi.v, (short)0, acc[s], false, false);
      acc[s] = __builtin_amdgcn_wmma_f32_16x16x32_bf16(
          false, a_hi, false, tb_lo.v, (short)0, acc[s], false, false);
      acc[s] = __builtin_amdgcn_wmma_f32_16x16x32_bf16(
          false, a_lo, false, tb_hi.v, (short)0, acc[s], false, false);
    }
    __syncthreads();
  }

  // ---- store C: 32-bit C/D 16x16 layout: VGPR v -> M = v + 8*hh, N = l16 ----
#pragma unroll
  for (int s = 0; s < 8; ++s) {
#pragma unroll
    for (int v = 0; v < 8; ++v) {
      const int m = v + 8 * hh;
      logits[(size_t)(rowBase + m) * NEXP + s * 16 + l16] = acc[s][v];
    }
  }
}

// ---------------------------------------------------------------------------
// Kernel 2: per-row deterministic top-6 (index tie-break) + softmax.
// One wave32 per row; 4 logits per lane.
// ---------------------------------------------------------------------------
__global__ __launch_bounds__(256)
void router_topk(const float* __restrict__ logits, float* __restrict__ idx_out,
                 float* __restrict__ w_out, int N) {
  const int lane = threadIdx.x & 31;
  const int row  = blockIdx.x * 8 + (threadIdx.x >> 5);
  if (row >= N) return;

  const float* lr = logits + (size_t)row * NEXP;
  float vals[4], adj[4];
  int   idxs[4];
#pragma unroll
  for (int j = 0; j < 4; ++j) {
    const int e = lane + 32 * j;
    const float x = lr[e];
    vals[j] = x;
    idxs[j] = e;
    adj[j]  = x - (float)e * 1e-9f;
  }

  float topv[TOPK];
  int   topi[TOPK];
#pragma unroll
  for (int r = 0; r < TOPK; ++r) {
    float ba = adj[0], bv = vals[0];
    int   bi = idxs[0];
#pragma unroll
    for (int j = 1; j < 4; ++j) {
      if (adj[j] > ba || (adj[j] == ba && idxs[j] < bi)) {
        ba = adj[j]; bv = vals[j]; bi = idxs[j];
      }
    }
    float ca = ba, cv = bv;
    int   ci = bi;
#pragma unroll
    for (int off = 16; off > 0; off >>= 1) {
      const float oa = __shfl_xor(ca, off, 32);
      const float ov = __shfl_xor(cv, off, 32);
      const int   oi = __shfl_xor(ci, off, 32);
      if (oa > ca || (oa == ca && oi < ci)) { ca = oa; cv = ov; ci = oi; }
    }
    topv[r] = cv;
    topi[r] = ci;
#pragma unroll
    for (int j = 0; j < 4; ++j)
      if (idxs[j] == ci) adj[j] = -3.4e38f;
  }

  float m = topv[0];
#pragma unroll
  for (int r = 1; r < TOPK; ++r) m = fmaxf(m, topv[r]);
  float ex[TOPK];
  float sum = 0.f;
#pragma unroll
  for (int r = 0; r < TOPK; ++r) { ex[r] = expf(topv[r] - m); sum += ex[r]; }
  const float inv = 1.0f / sum;

#pragma unroll
  for (int r = 0; r < TOPK; ++r) {
    if (lane == r) {
      idx_out[(size_t)row * TOPK + r] = (float)topi[r];
      w_out[(size_t)row * TOPK + r]   = ex[r] * inv;
    }
  }
}

// ---------------------------------------------------------------------------
extern "C" void kernel_launch(void* const* d_in, const int* in_sizes, int n_in,
                              void* d_out, int out_size, void* d_ws, size_t ws_size,
                              hipStream_t stream) {
  const float* X = (const float*)d_in[0];   // [B,S,H] fp32
  const float* W = (const float*)d_in[1];   // [E,H]   fp32
  const int N = in_sizes[0] / HID;          // B*S tokens

  float* logits  = (float*)d_out;
  float* idx_out = logits + (size_t)N * NEXP;
  float* w_out   = idx_out + (size_t)N * TOPK;

  const size_t need = 2 * (size_t)NEXP * HID * sizeof(__bf16);  // 1 MB
  if (ws_size >= need) {
    __bf16* Whi = (__bf16*)d_ws;
    __bf16* Wlo = Whi + (size_t)NEXP * HID;
    split_gate<<<(NEXP * HID) / (256 * 8), 256, 0, stream>>>(W, Whi, Wlo);
    router_gemm<true><<<N / ROWS_PER_BLOCK, 256, 0, stream>>>(X, W, Whi, Wlo, logits);
  } else {
    router_gemm<false><<<N / ROWS_PER_BLOCK, 256, 0, stream>>>(X, W, nullptr, nullptr, logits);
  }
  router_topk<<<(N + 7) / 8, 256, 0, stream>>>(logits, idx_out, w_out, N);
}
